// PatchNetExtended_77756087926784
// MI455X (gfx1250) — matrix operand order repaired and checked by reading, not compiled, not run on hardware
//
#include <hip/hip_runtime.h>
#include <hip/hip_bf16.h>
#include <math.h>

// ---------------------------------------------------------------------------
// PatchNet forward for MI455X (gfx1250, wave32).
// ~35 GFLOP of short-K GEMMs; all tensors (tables 8MB, staged bf16
// activations 42MB) are L2-resident (192MB), so HBM (23.3 TB/s) is irrelevant
// and throughput is set by the WMMA pipe. Convs run as implicit GEMM on
// v_wmma_f32_16x16x32_bf16 (f32 accumulate; contractions <= 5760, inputs
// ~N(0,0.05^2) -> bf16-safe). The dominant line conv stages its x-tile and W
// in LDS via CDNA5 async copies (global_load_async_to_lds_b128 +
// s_wait_asynccnt), with bank-conflict-free padded strides, so the hot loop
// is ds_load + wmma instead of 64-bit-addressed global loads.
// ---------------------------------------------------------------------------

typedef __attribute__((ext_vector_type(16))) __bf16 v16bf;
typedef __attribute__((ext_vector_type(8)))  float  v8f;

union FragBF { v16bf v; unsigned int u[8]; };

__device__ __forceinline__ unsigned lds_addr_of(const void* p) {
  // LDS generic addresses: high bits = shared aperture, low 32 = LDS offset.
  return (unsigned)(reinterpret_cast<uintptr_t>(p));
}

// CDNA5 async global->LDS copy (ASYNCcnt-tracked DMA path, portable inline asm)
__device__ __forceinline__ void async_copy_b128(unsigned ldsOff, const void* g) {
  asm volatile("global_load_async_to_lds_b128 %0, %1, off"
               :: "v"(ldsOff), "v"((unsigned long long)(uintptr_t)g)
               : "memory");
}
__device__ __forceinline__ void async_wait0() {
  asm volatile("s_wait_asynccnt 0" ::: "memory");
}

// ---- fp32 -> bf16 elementwise (weights) -----------------------------------
__global__ void f32_to_bf16_kernel(const float* __restrict__ in,
                                   __bf16* __restrict__ out, int n) {
  int i = blockIdx.x * blockDim.x + threadIdx.x;
  if (i < n) out[i] = (__bf16)in[i];
}

// ---- embedding gather fused with bf16 convert (D = 64) --------------------
__global__ void gather_embed_bf16_kernel(const int* __restrict__ tok,
                                         const float* __restrict__ tbl,
                                         __bf16* __restrict__ out,
                                         long long nElem) {
  long long i = (long long)blockIdx.x * blockDim.x + threadIdx.x;
  if (i >= nElem) return;
  int t = tok[i >> 6];
  out[i] = (__bf16)tbl[((long long)t << 6) + (i & 63)];
}

// ---- scatter ftr into concat buffer cols [0,32) ---------------------------
__global__ void copy_ftr_kernel(const float* __restrict__ ftr,
                                float* __restrict__ xcat, int n) {
  int i = blockIdx.x * blockDim.x + threadIdx.x;
  if (i < n) xcat[(i >> 5) * 992 + (i & 31)] = ftr[i];
}

// ---------------------------------------------------------------------------
// Fused conv(+bias)+ReLU+max-over-time as WMMA implicit GEMM.
//   x: [Nseq, L, D] bf16, w: [64, K*D] bf16 (k = tap*D + d), bias: [64] f32.
// One wave per sequence; M = 16 time steps/tile, N = Co=64 (4 accumulators),
// K-dim = K*D in bf16 steps of 32.
// STAGE_X: per-wave async-copy of the sequence into LDS (requires L<=32,D=64).
// STAGE_W: block-cooperative async-copy of W into LDS (requires K*D<=192).
// Out index for seq s, ch c: (s/groups)*rowStride + colOffset +
//                            (s%groups)*groupStride + c
// ---------------------------------------------------------------------------
template <bool STAGE_X, bool STAGE_W>
__global__ __launch_bounds__(256)
void wmma_conv_relu_max_kernel(const __bf16* __restrict__ x,
                               const __bf16* __restrict__ w,
                               const float* __restrict__ bias,
                               float* __restrict__ outF,
                               __bf16* __restrict__ outB,
                               int Nseq, int L, int D, int K,
                               int groups, int rowStride, int colOffset,
                               int groupStride) {
  const int lane = threadIdx.x & 31;
  const int wav  = threadIdx.x >> 5;
  const int seq  = blockIdx.x * 8 + wav;

  const int T    = L - K + 1;
  const int KD   = K * D;
  const int nk   = KD >> 5;              // k-steps of 32 bf16
  const int m    = lane & 15;            // A-row / B-column fed by this lane
  const int half = lane >> 4;            // K-subrange selector

  // ---- optional W staging: rows padded to KD+8 elems -> the 16 B-lanes
  //      land on 16 distinct banks (row stride = 36/4/36 mod 64 banks). ----
  const int WS = KD + 8;
  __bf16* wl = nullptr;
  if constexpr (STAGE_W) {
    __shared__ __bf16 smw[64 * 200];     // max 64*(192+8) = 25.0 KB
    const int row = threadIdx.x >> 2, q = threadIdx.x & 3;
    const int chunks = KD >> 3;          // 16B chunks per W row
    unsigned wbase = lds_addr_of(smw) + (unsigned)(row * WS * 2);
    const char* gsrc = (const char*)w + (size_t)row * KD * 2;
    for (int cc = q; cc < chunks; cc += 4)
      async_copy_b128(wbase + cc * 16, gsrc + cc * 16);
    async_wait0();
    __syncthreads();
    wl = smw;
  }

  if (seq >= Nseq) return;               // uniform per wave (after barrier)

  const __bf16* __restrict__ xs = x + (long long)seq * L * D;

  // ---- optional x staging: rows padded to 72 elems (144B) -> A-lane rows
  //      map to banks m*36 mod 64 = all-distinct for m = 0..15. ----
  const int XS = 72;
  __bf16* xl = nullptr;
  if constexpr (STAGE_X) {
    __shared__ __bf16 smx[8 * 32 * 72];  // 8 waves * 32 rows * 144B = 36 KB
    __bf16* mine = smx + wav * 32 * XS;
    unsigned xbase = lds_addr_of(mine);
    const int rowChunks = (D * 2) >> 4;  // 8 for D=64
    for (int r = lane; r < L; r += 32) {
      unsigned rb = xbase + (unsigned)(r * XS * 2);
      const char* gs = (const char*)(xs + r * D);
      for (int cc = 0; cc < rowChunks; ++cc)
        async_copy_b128(rb + cc * 16, gs + cc * 16);
    }
    async_wait0();                       // per-wave; no barrier needed
    xl = mine;
  }

  const v8f vzero = {0.f, 0.f, 0.f, 0.f, 0.f, 0.f, 0.f, 0.f};
  float runmax[4] = {-3.0e38f, -3.0e38f, -3.0e38f, -3.0e38f};

  for (int t0 = 0; t0 < T; t0 += 16) {
    v8f acc[4];
#pragma unroll
    for (int nt = 0; nt < 4; ++nt) acc[nt] = vzero;

    for (int kk = 0; kk < nk; ++kk) {
      const int kbase = kk << 5;
      // ---- A fragment (ISA layout: lane=M, VGPR j holds k-pair at
      //      (j>>2)*16 + half*8 + (j&3)*2) ----
      FragBF a;
#pragma unroll
      for (int j = 0; j < 8; ++j) {
        int k0  = kbase + ((j >> 2) << 4) + (half << 3) + ((j & 3) << 1);
        int tap = (int)(k0 >= D) + (int)(k0 >= 2 * D);   // K <= 3
        int d0  = k0 - tap * D;                           // even -> dword
        int trow = t0 + m + tap;
        trow = trow < L ? trow : L - 1;  // clamp OOB rows; masked at reduce
        if constexpr (STAGE_X)
          a.u[j] = *reinterpret_cast<const unsigned int*>(xl + trow * XS + d0);
        else
          a.u[j] = *reinterpret_cast<const unsigned int*>(xs + trow * D + d0);
      }
      // ---- 4 N-tiles of B (lane=N column, k-pair at (j>>2)*16 + half*8 + (j&3)*2) ----
#pragma unroll
      for (int nt = 0; nt < 4; ++nt) {
        FragBF b;
        if constexpr (STAGE_W) {
          const __bf16* wr = wl + (nt * 16 + m) * WS + kbase;
#pragma unroll
          for (int j = 0; j < 8; ++j)
            b.u[j] = *reinterpret_cast<const unsigned int*>(wr + ((j >> 2) << 4) + (half << 3) + ((j & 3) << 1));
        } else {
          const __bf16* wr = w + (nt * 16 + m) * KD + kbase;
#pragma unroll
          for (int j = 0; j < 8; ++j)
            b.u[j] = *reinterpret_cast<const unsigned int*>(wr + ((j >> 2) << 4) + (half << 3) + ((j & 3) << 1));
        }
        acc[nt] = __builtin_amdgcn_wmma_f32_16x16x32_bf16(
            false, a.v, false, b.v, (short)0, acc[nt], false, false);
      }
    }
    // ---- in-tile max over time rows (C/D layout: M = r + half*8) ----------
#pragma unroll
    for (int nt = 0; nt < 4; ++nt) {
#pragma unroll
      for (int r = 0; r < 8; ++r) {
        int mt = t0 + (half << 3) + r;
        if (mt < T) runmax[nt] = fmaxf(runmax[nt], acc[nt][r]);
      }
    }
  }

  // combine the two lane-halves (same channel, disjoint M rows) — wave32 xor16
  const long long r0 = (long long)(seq / groups) * rowStride + colOffset +
                       (long long)(seq % groups) * groupStride;
#pragma unroll
  for (int nt = 0; nt < 4; ++nt) {
    float v = fmaxf(runmax[nt], __shfl_xor(runmax[nt], 16, 32));
    if (lane < 16) {
      int ch  = nt * 16 + lane;
      float y = v + bias[ch];
      y = y > 0.0f ? y : 0.0f;                 // relu(max) == max(relu)
      if (outF) outF[r0 + ch] = y;
      else      outB[r0 + ch] = (__bf16)y;
    }
  }
}

// ---- fc1: h = relu(xcat @ fc1_w^T + b), [64,992]x[512,992] ----------------
__global__ void fc1_kernel(const float* __restrict__ x,
                           const float* __restrict__ w,
                           const float* __restrict__ b,
                           float* __restrict__ h) {
  int idx = blockIdx.x * blockDim.x + threadIdx.x;
  if (idx >= 64 * 512) return;
  int bb = idx >> 9, j = idx & 511;
  const float* xr = x + bb * 992;
  const float* wr = w + j * 992;
  float s = 0.f;
#pragma unroll 4
  for (int k = 0; k < 992; ++k) s += xr[k] * wr[k];
  s += b[j];
  h[idx] = s > 0.f ? s : 0.f;
}

// ---- fc2 + sigmoid --------------------------------------------------------
__global__ void fc2_kernel(const float* __restrict__ h,
                           const float* __restrict__ w,
                           const float* __restrict__ b,
                           float* __restrict__ out) {
  int bb = threadIdx.x;
  if (bb >= 64) return;
  float s = 0.f;
#pragma unroll 4
  for (int k = 0; k < 512; ++k) s += h[bb * 512 + k] * w[k];
  s += b[0];
  out[bb] = 1.f / (1.f + __expf(-s));
}

// ---------------------------------------------------------------------------
extern "C" void kernel_launch(void* const* d_in, const int* in_sizes, int n_in,
                              void* d_out, int out_size, void* d_ws, size_t ws_size,
                              hipStream_t stream) {
  (void)in_sizes; (void)n_in; (void)out_size; (void)ws_size;

  constexpr int B = 64, Lm = 256, F = 2, H = 8, CL = 10, Wd = 32, D = 64;
  constexpr int CoK = 192, HID = 512, FCIN = 992, FTR = 32;
  constexpr int NLINE = B * F * H * CL;     // 10240 lines
  constexpr int NH = B * F;                 // 128 hunk sequences
  constexpr int D2 = CL * CoK;              // 1920 hunk feature dim

  // ---- inputs (setup_inputs order; tuples flattened element-wise) ----
  const float* ftr   = (const float*)d_in[0];
  const int*   msg   = (const int*)  d_in[1];
  const int*   addc  = (const int*)  d_in[2];
  const int*   remc  = (const int*)  d_in[3];
  const float* embM  = (const float*)d_in[4];
  const float* embC  = (const float*)d_in[5];
  const float* w_msg[3]  = {(const float*)d_in[6],  (const float*)d_in[7],  (const float*)d_in[8]};
  const float* b_msg[3]  = {(const float*)d_in[9],  (const float*)d_in[10], (const float*)d_in[11]};
  const float* w_line[3] = {(const float*)d_in[12], (const float*)d_in[13], (const float*)d_in[14]};
  const float* b_line[3] = {(const float*)d_in[15], (const float*)d_in[16], (const float*)d_in[17]};
  const float* w_hunk[3] = {(const float*)d_in[18], (const float*)d_in[19], (const float*)d_in[20]};
  const float* b_hunk[3] = {(const float*)d_in[21], (const float*)d_in[22], (const float*)d_in[23]};
  const float* fc1w = (const float*)d_in[24];
  const float* fc1b = (const float*)d_in[25];
  const float* fc2w = (const float*)d_in[26];
  const float* fc2b = (const float*)d_in[27];

  // ---- workspace carve (~50 MB; codeX/lineF reused for added then removed)
  char* p = (char*)d_ws;
  auto carve = [&](size_t bytes) -> char* {
    char* r = p; p += (bytes + 255) & ~(size_t)255; return r;
  };
  __bf16* msgX   = (__bf16*)carve((size_t)B * Lm * D * 2);        // 2 MB
  __bf16* codeX  = (__bf16*)carve((size_t)NLINE * Wd * D * 2);    // 42 MB
  __bf16* lineF  = (__bf16*)carve((size_t)NLINE * CoK * 2);       // 3.9 MB
  __bf16* wMsgB  = (__bf16*)carve((size_t)24576 * 2);
  __bf16* wLineB = (__bf16*)carve((size_t)24576 * 2);
  __bf16* wHunkB = (__bf16*)carve((size_t)737280 * 2);
  float*  xcat   = (float*)carve((size_t)B * FCIN * 4);
  float*  hbuf   = (float*)carve((size_t)B * HID * 4);

  const int smallWoff[3] = {0, 4096, 12288};               // K=1,2,3 (D=64)
  const int smallWsz[3]  = {4096, 8192, 12288};
  const int hunkWoff[3]  = {0, 122880, 368640};            // K=1,2,3 (D=1920)
  const int hunkWsz[3]   = {122880, 245760, 368640};

  const dim3 blk(256);
  auto grid1 = [](long long n) { return dim3((unsigned)((n + 255) / 256)); };

  // ---- 1) weights -> bf16 ----
  for (int k = 0; k < 3; ++k) {
    f32_to_bf16_kernel<<<grid1(smallWsz[k]), blk, 0, stream>>>(w_msg[k],  wMsgB  + smallWoff[k], smallWsz[k]);
    f32_to_bf16_kernel<<<grid1(smallWsz[k]), blk, 0, stream>>>(w_line[k], wLineB + smallWoff[k], smallWsz[k]);
    f32_to_bf16_kernel<<<grid1(hunkWsz[k]),  blk, 0, stream>>>(w_hunk[k], wHunkB + hunkWoff[k],  hunkWsz[k]);
  }

  // ---- 2) ftr into concat cols [0,32) ----
  copy_ftr_kernel<<<grid1(B * FTR), blk, 0, stream>>>(ftr, xcat, B * FTR);

  // ---- 3) msg path: gather -> 3 convs into xcat cols [32,224) ----
  gather_embed_bf16_kernel<<<grid1((long long)B * Lm * D), blk, 0, stream>>>(
      msg, embM, msgX, (long long)B * Lm * D);
  for (int k = 0; k < 3; ++k)
    wmma_conv_relu_max_kernel<false, true><<<dim3((B + 7) / 8), blk, 0, stream>>>(
        msgX, wMsgB + smallWoff[k], b_msg[k], xcat, nullptr,
        B, Lm, D, k + 1, /*groups*/1, FCIN, FTR + 64 * k, 0);

  // ---- 4) code paths: added -> xcat [224,608), removed -> [608,992) ----
  const int* codes[2] = {addc, remc};
  const int  cbase[2] = {224, 608};
  for (int s = 0; s < 2; ++s) {
    gather_embed_bf16_kernel<<<grid1((long long)NLINE * Wd * D), blk, 0, stream>>>(
        codes[s], embC, codeX, (long long)NLINE * Wd * D);
    // line-level TextCNN (dominant ~15.4 GFLOP): x+W staged in LDS via async
    for (int k = 0; k < 3; ++k)
      wmma_conv_relu_max_kernel<true, true><<<dim3((NLINE + 7) / 8), blk, 0, stream>>>(
          codeX, wLineB + smallWoff[k], b_line[k], nullptr, lineF,
          NLINE, Wd, D, k + 1, /*groups*/1, CoK, 64 * k, 0);
    // hunk-level conv over H=8 with 1920-wide features; W too big for LDS
    for (int k = 0; k < 3; ++k)
      wmma_conv_relu_max_kernel<false, false><<<dim3((NH + 7) / 8), blk, 0, stream>>>(
          lineF, wHunkB + hunkWoff[k], b_hunk[k], xcat, nullptr,
          NH, H, D2, k + 1, /*groups*/F, FCIN, cbase[s] + 64 * k, CoK);
  }

  // ---- 5) MLP head ----
  fc1_kernel<<<grid1(B * HID), blk, 0, stream>>>(xcat, fc1w, fc1b, hbuf);
  fc2_kernel<<<dim3(1), dim3(64), 0, stream>>>(hbuf, fc2w, fc2b, (float*)d_out);
}
